// GraphMamba_68281390071815
// MI455X (gfx1250) — compile-verified, matrix operand order checked by reading, not run on hardware
//
#include <hip/hip_runtime.h>
#include <hip/hip_bf16.h>
#include <math.h>

// ---------------------------------------------------------------------------
// Types for CDNA5 WMMA (wave32): v_wmma_f32_16x16x32_bf16
// ---------------------------------------------------------------------------
typedef __bf16 bf16_t;
typedef __attribute__((ext_vector_type(16))) __bf16 v16bf;
typedef __attribute__((ext_vector_type(8)))  __bf16 v8bf;
typedef __attribute__((ext_vector_type(8)))  float  v8f;

#define BM 128
#define BN 64
#define BK 32
#define LDSK 40   // padded K stride (bf16 units): 80B rows, 16B aligned, conflict-free b128 reads

// Problem constants
#define Bb 2
#define Tt 64
#define Nn 512
#define Ee 1024
#define Hh 4
#define Cc 64
#define DM 256
#define DI 512
#define DSs 16
#define DR 16
#define KK 4
#define ROWS 65536            // B*T*N == B*N*T
#define ETOT 1536             // E + N self loops
#define GG 128                // B*T graphs

// ---------------------------------------------------------------------------
// Fragment load: elements 0..7 = K[half*8 .. +7], 8..15 = K[16+half*8 .. +7]
// (ISA 7.12.2 16-bit A 16x32 layout; B is loaded column-major with same math)
// ---------------------------------------------------------------------------
__device__ __forceinline__ v16bf frag_from_lds(const bf16_t* p) {
    v8bf lo = *reinterpret_cast<const v8bf*>(p);
    v8bf hi = *reinterpret_cast<const v8bf*>(p + 16);
    return __builtin_shufflevector(lo, hi, 0,1,2,3,4,5,6,7,8,9,10,11,12,13,14,15);
}

// ---------------------------------------------------------------------------
// Tile staging: batch ALL global loads into registers first (full MLP of the
// memory pipe, single wait), then convert+store to LDS.
// FULL=false uses clamped in-allocation addresses + select-0 (no exec dance).
// ---------------------------------------------------------------------------
template<bool FULL>
__device__ __forceinline__ void stage_load(
    const float* __restrict__ A, int lda,
    const float* __restrict__ Bw, int ldb,
    int m0, int n0, int k0, int M, int N, int K,
    int tid, int bn, int bkk,
    float4 (&ar)[4], float (&br)[8])
{
    #pragma unroll
    for (int pass = 0; pass < 4; ++pass) {
        int p   = pass * 1024 + tid * 4;
        int row = p >> 5, col = p & 31;
        if (FULL) {
            ar[pass] = *reinterpret_cast<const float4*>(A + (size_t)(m0 + row) * lda + k0 + col);
        } else {
            int gr = m0 + row;
            int cr = gr < M ? gr : (M - 1);
            int gc = k0 + col;
            const float* ap = A + (size_t)cr * lda;
            float x0 = ap[gc + 0 < K ? gc + 0 : K - 1];
            float x1 = ap[gc + 1 < K ? gc + 1 : K - 1];
            float x2 = ap[gc + 2 < K ? gc + 2 : K - 1];
            float x3 = ap[gc + 3 < K ? gc + 3 : K - 1];
            bool okr = gr < M;
            ar[pass].x = (okr && gc + 0 < K) ? x0 : 0.f;
            ar[pass].y = (okr && gc + 1 < K) ? x1 : 0.f;
            ar[pass].z = (okr && gc + 2 < K) ? x2 : 0.f;
            ar[pass].w = (okr && gc + 3 < K) ? x3 : 0.f;
        }
    }
    #pragma unroll
    for (int j = 0; j < 8; ++j) {
        if (FULL) {
            br[j] = Bw[(size_t)(k0 + bkk + j) * ldb + (n0 + bn)];
        } else {
            int gk = k0 + bkk + j, gn = n0 + bn;
            float v = Bw[(size_t)(gk < K ? gk : K - 1) * ldb + (gn < N ? gn : N - 1)];
            br[j] = (gk < K && gn < N) ? v : 0.f;
        }
    }
}

__device__ __forceinline__ void stage_store(
    bf16_t (&As)[BM][LDSK], bf16_t (&Bs)[BN][LDSK],
    int tid, int bn, int bkk,
    const float4 (&ar)[4], const float (&br)[8])
{
    #pragma unroll
    for (int pass = 0; pass < 4; ++pass) {
        int p   = pass * 1024 + tid * 4;
        int row = p >> 5, col = p & 31;
        As[row][col + 0] = (bf16_t)ar[pass].x;
        As[row][col + 1] = (bf16_t)ar[pass].y;
        As[row][col + 2] = (bf16_t)ar[pass].z;
        As[row][col + 3] = (bf16_t)ar[pass].w;
    }
    #pragma unroll
    for (int j = 0; j < 8; ++j)
        Bs[bn][bkk + j] = (bf16_t)br[j];
}

// ---------------------------------------------------------------------------
// GEMM: C[M,N] (=|+=) act(A[M,K] @ B[K,N] + bias), fp32 in/out, bf16 WMMA,
// two-stage software pipeline: next tile's global loads issue under the
// current tile's WMMAs.
// ---------------------------------------------------------------------------
template<bool ACCUM, int ACT, bool FULL>
__global__ __launch_bounds__(256) void gemm_wmma_bf16(
    const float* __restrict__ A, int lda,
    const float* __restrict__ Bw, int ldb,
    const float* __restrict__ bias,
    float* __restrict__ C, int ldc,
    int M, int N, int K)
{
    __shared__ alignas(16) bf16_t As[BM][LDSK];
    __shared__ alignas(16) bf16_t Bs[BN][LDSK];

    const int tid  = threadIdx.x;
    const int lane = tid & 31;
    const int wave = tid >> 5;
    const int wm = wave >> 1;          // 0..3 (M)
    const int wn = wave & 1;           // 0..1 (N)
    const int m0 = blockIdx.x * BM;
    const int n0 = blockIdx.y * BN;
    const int r    = lane & 15;
    const int half = lane >> 4;
    const int bn  = tid >> 2;          // B tile row (n)
    const int bkk = (tid & 3) * 8;     // B tile k base

    const v8f zf = {0.f,0.f,0.f,0.f,0.f,0.f,0.f,0.f};
    v8f acc[2][2];
    acc[0][0] = zf; acc[0][1] = zf; acc[1][0] = zf; acc[1][1] = zf;

    float4 ar[4];
    float  br[8];
    stage_load<FULL>(A, lda, Bw, ldb, m0, n0, 0, M, N, K, tid, bn, bkk, ar, br);

    for (int k0 = 0; k0 < K; k0 += BK) {
        stage_store(As, Bs, tid, bn, bkk, ar, br);
        __syncthreads();
        if (k0 + BK < K)   // issue next tile's loads; latency hides under WMMAs
            stage_load<FULL>(A, lda, Bw, ldb, m0, n0, k0 + BK, M, N, K, tid, bn, bkk, ar, br);

        v16bf af0 = frag_from_lds(&As[wm * 32 +  0 + r][half * 8]);
        v16bf af1 = frag_from_lds(&As[wm * 32 + 16 + r][half * 8]);
        v16bf bf0 = frag_from_lds(&Bs[wn * 32 +  0 + r][half * 8]);
        v16bf bf1 = frag_from_lds(&Bs[wn * 32 + 16 + r][half * 8]);

        acc[0][0] = __builtin_amdgcn_wmma_f32_16x16x32_bf16(false, af0, false, bf0, (short)0, acc[0][0], false, false);
        acc[0][1] = __builtin_amdgcn_wmma_f32_16x16x32_bf16(false, af0, false, bf1, (short)0, acc[0][1], false, false);
        acc[1][0] = __builtin_amdgcn_wmma_f32_16x16x32_bf16(false, af1, false, bf0, (short)0, acc[1][0], false, false);
        acc[1][1] = __builtin_amdgcn_wmma_f32_16x16x32_bf16(false, af1, false, bf1, (short)0, acc[1][1], false, false);
        __syncthreads();
    }

    // ---- epilogue: C/D layout row = half*8 + e, col = lane&15
    #pragma unroll
    for (int i = 0; i < 2; ++i)
    #pragma unroll
    for (int j = 0; j < 2; ++j)
    #pragma unroll
    for (int e = 0; e < 8; ++e) {
        int row = m0 + wm * 32 + i * 16 + half * 8 + e;
        int col = n0 + wn * 32 + j * 16 + r;
        if (FULL || (row < M && col < N)) {
            float v = acc[i][j][e];
            if (bias) v += bias[col];
            if (ACT == 1) v = v > 0.f ? v : 0.f;
            if (ACCUM) C[(size_t)row * ldc + col] += v;
            else       C[(size_t)row * ldc + col]  = v;
        }
    }
}

// ---------------------------------------------------------------------------
// Elementwise / graph / scan kernels
// ---------------------------------------------------------------------------
__device__ __forceinline__ void atomicMaxF(float* addr, float value) {
    if (value >= 0.f) atomicMax((int*)addr, __float_as_int(value));
    else              atomicMin((unsigned int*)addr, __float_as_uint(value));
}

__global__ void k_input_proj(const float* __restrict__ vmag, const float* __restrict__ pbus,
                             const float* __restrict__ qbus, const float* __restrict__ mask,
                             const float* __restrict__ w_in, const float* __restrict__ b_in,
                             float* __restrict__ X) {
    int idx = blockIdx.x * blockDim.x + threadIdx.x;
    if (idx >= ROWS * Cc) return;
    int c = idx & 63, row = idx >> 6;
    float m = mask[row];
    float a = vmag[row] * m, p = pbus[row] * m, q = qbus[row] * m;
    X[idx] = a * w_in[c] + p * w_in[64 + c] + q * w_in[128 + c] + b_in[c];
}

__global__ void k_nodesum(const float* __restrict__ XH, const float* __restrict__ asrc,
                          const float* __restrict__ adst, float* __restrict__ ES, float* __restrict__ ED) {
    int idx = blockIdx.x * blockDim.x + threadIdx.x;
    if (idx >= ROWS * Hh) return;
    int h = idx & 3, row = idx >> 2;
    const float* x = XH + (size_t)row * DM + h * Cc;
    float s = 0.f, d = 0.f;
    for (int c = 0; c < Cc; ++c) { float v = x[c]; s += v * asrc[h * Cc + c]; d += v * adst[h * Cc + c]; }
    ES[idx] = s; ED[idx] = d;
}

__global__ void k_fill(float* __restrict__ p, float v, int n) {
    int idx = blockIdx.x * blockDim.x + threadIdx.x;
    if (idx < n) p[idx] = v;
}

__global__ void k_edge_max(const int* __restrict__ ei, const float* __restrict__ ES,
                           const float* __restrict__ ED, float* __restrict__ EATT, float* __restrict__ MM) {
    int idx = blockIdx.x * blockDim.x + threadIdx.x;
    if (idx >= GG * ETOT * Hh) return;
    int h = idx & 3, ge = idx >> 2;
    int e = ge % ETOT, g = ge / ETOT;
    int s = (e < Ee) ? ei[e] : (e - Ee);
    int d = (e < Ee) ? ei[Ee + e] : (e - Ee);
    float v = ES[((size_t)g * Nn + s) * Hh + h] + ED[((size_t)g * Nn + d) * Hh + h];
    v = v < 0.f ? 0.2f * v : v;
    EATT[idx] = v;
    atomicMaxF(&MM[((size_t)g * Nn + d) * Hh + h], v);
}

__global__ void k_edge_expsum(const int* __restrict__ ei, float* __restrict__ EATT,
                              const float* __restrict__ MM, float* __restrict__ DEN) {
    int idx = blockIdx.x * blockDim.x + threadIdx.x;
    if (idx >= GG * ETOT * Hh) return;
    int h = idx & 3, ge = idx >> 2;
    int e = ge % ETOT, g = ge / ETOT;
    int d = (e < Ee) ? ei[Ee + e] : (e - Ee);
    float w = __expf(EATT[idx] - MM[((size_t)g * Nn + d) * Hh + h]);
    EATT[idx] = w;
    atomicAdd(&DEN[((size_t)g * Nn + d) * Hh + h], w);
}

__global__ void k_init_bias256(float* __restrict__ out, const float* __restrict__ bias, int n) {
    int idx = blockIdx.x * blockDim.x + threadIdx.x;
    if (idx < n) out[idx] = bias[idx & 255];
}

__global__ void k_edge_agg(const int* __restrict__ ei, const float* __restrict__ EATT,
                           const float* __restrict__ DEN, const float* __restrict__ XH,
                           float* __restrict__ OUT) {
    int ge = blockIdx.x;                  // g*ETOT + e
    int g = ge / ETOT, e = ge % ETOT;
    int hc = threadIdx.x;                 // h*64 + c
    int h = hc >> 6;
    int s = (e < Ee) ? ei[e] : (e - Ee);
    int d = (e < Ee) ? ei[Ee + e] : (e - Ee);
    float a = EATT[(size_t)ge * Hh + h] / DEN[((size_t)g * Nn + d) * Hh + h];
    atomicAdd(&OUT[((size_t)g * Nn + d) * DM + hc], a * XH[((size_t)g * Nn + s) * DM + hc]);
}

__global__ void k_elu(float* __restrict__ p, int n) {
    int idx = blockIdx.x * blockDim.x + threadIdx.x;
    if (idx >= n) return;
    float v = p[idx];
    p[idx] = v > 0.f ? v : (__expf(v) - 1.f);
}

__global__ void k_transpose(const float* __restrict__ HS, float* __restrict__ U) {
    int idx = blockIdx.x * blockDim.x + threadIdx.x;
    if (idx >= ROWS * DM) return;
    int dd = idx & 255; int row = idx >> 8;          // row = (b*T+t)*N + n
    int n = row & 511, bt = row >> 9;
    int t = bt & 63, b = bt >> 6;
    U[(((size_t)(b * Nn + n) * Tt + t) << 8) + dd] = HS[idx];
}

__global__ void k_layernorm(const float* __restrict__ in, float* __restrict__ out,
                            const float* __restrict__ g, const float* __restrict__ b, int D) {
    extern __shared__ float red[];
    int row = blockIdx.x, tid = threadIdx.x;
    float v = in[(size_t)row * D + tid];
    red[tid] = v; __syncthreads();
    for (int s = D >> 1; s > 0; s >>= 1) { if (tid < s) red[tid] += red[tid + s]; __syncthreads(); }
    float mean = red[0] / (float)D; __syncthreads();
    float dv = v - mean;
    red[tid] = dv * dv; __syncthreads();
    for (int s = D >> 1; s > 0; s >>= 1) { if (tid < s) red[tid] += red[tid + s]; __syncthreads(); }
    float var = red[0] / (float)D;
    out[(size_t)row * D + tid] = dv * rsqrtf(var + 1e-5f) * g[tid] + b[tid];
}

__global__ void k_conv_silu(const float* __restrict__ XP, const float* __restrict__ convw,
                            const float* __restrict__ convb, float* __restrict__ XM) {
    int idx = blockIdx.x * blockDim.x + threadIdx.x;
    if (idx >= ROWS * DI) return;
    int d = idx & 511, t = (idx >> 9) & 63, s = idx >> 15;
    const float* w = convw + d * KK;
    float acc = convb[d];
    #pragma unroll
    for (int k = 0; k < KK; ++k) {
        int tt = t - (KK - 1) + k;
        if (tt >= 0) acc += XP[(((size_t)s * Tt + tt) << 9) + d] * w[k];
    }
    XM[idx] = acc / (1.f + __expf(-acc));  // silu
}

__global__ void k_softplus_bias(float* __restrict__ P, const float* __restrict__ dtb, int n) {
    int idx = blockIdx.x * blockDim.x + threadIdx.x;
    if (idx >= n) return;
    float x = P[idx] + dtb[idx & 511];
    P[idx] = (x > 20.f) ? x : log1pf(__expf(x));
}

__global__ __launch_bounds__(256) void k_scan(const float* __restrict__ DT, const float* __restrict__ XD,
                                              float* __restrict__ XMY, const float* __restrict__ Z,
                                              const float* __restrict__ Alog, const float* __restrict__ Dp) {
    __shared__ float Bs[DSs], Cs[DSs];
    int s = blockIdx.x >> 1;
    int d = ((blockIdx.x & 1) << 8) + threadIdx.x;
    float Ad[DSs], h[DSs];
    #pragma unroll
    for (int j = 0; j < DSs; ++j) { Ad[j] = -__expf(Alog[(size_t)d * DSs + j]); h[j] = 0.f; }
    float Dd = Dp[d];
    for (int t = 0; t < Tt; ++t) {
        size_t rb = (size_t)s * Tt + t;
        if (threadIdx.x < 32) {
            float v = XD[rb * 48 + 16 + threadIdx.x];
            if (threadIdx.x < 16) Bs[threadIdx.x] = v; else Cs[threadIdx.x - 16] = v;
        }
        __syncthreads();
        size_t o = (rb << 9) + d;
        float dt = DT[o], x = XMY[o], z = Z[o];
        float dx = dt * x, acc = 0.f;
        #pragma unroll
        for (int j = 0; j < DSs; ++j) {
            h[j] = __expf(dt * Ad[j]) * h[j] + dx * Bs[j];
            acc += h[j] * Cs[j];
        }
        float y = acc + Dd * x;
        y *= z / (1.f + __expf(-z));       // * silu(z)
        XMY[o] = y;
        __syncthreads();
    }
}

__global__ void k_pool(const int* __restrict__ ei, const float* __restrict__ U, float* __restrict__ POOL) {
    int idx = blockIdx.x * blockDim.x + threadIdx.x;
    if (idx >= Bb * Ee * 2 * DM) return;
    int f = idx & 511, e = (idx >> 9) & 1023, b = idx >> 19;
    int node = (f < DM) ? ei[e] : ei[Ee + e];
    int dd = f & 255;
    float inv = (1.f - 0.9f) / (1.f - powf(0.9f, (float)Tt));
    float w = inv, acc = 0.f;
    const float* base = U + (((size_t)(b * Nn + node) * Tt) << 8) + dd;
    for (int t = Tt - 1; t >= 0; --t) { acc += w * base[(size_t)t << 8]; w *= 0.9f; }
    POOL[idx] = acc;
}

__global__ void k_head_sh(const float* __restrict__ X, const float* __restrict__ w3,
                          const float* __restrict__ b3, float* __restrict__ out) {
    int r = blockIdx.x * blockDim.x + threadIdx.x;
    if (r >= Bb * Nn) return;
    float s0 = b3[0], s1 = b3[1];
    for (int j = 0; j < 64; ++j) { float v = X[(size_t)r * 64 + j]; s0 += v * w3[j * 2]; s1 += v * w3[j * 2 + 1]; }
    int b = r >> 9, n = r & 511;
    out[(size_t)b * 3072 + n]       = 0.3f / (1.f + __expf(-s0)) + 0.85f;
    out[(size_t)b * 3072 + 512 + n] = tanhf(s1) * 0.5f;
}

__global__ void k_head_ph(const float* __restrict__ X, const float* __restrict__ w3,
                          const float* __restrict__ b3, float* __restrict__ out) {
    int r = blockIdx.x * blockDim.x + threadIdx.x;
    if (r >= Bb * Ee) return;
    float s0 = b3[0], s1 = b3[1];
    for (int j = 0; j < 64; ++j) { float v = X[(size_t)r * 64 + j]; s0 += v * w3[j * 2]; s1 += v * w3[j * 2 + 1]; }
    int b = r >> 10, e = r & 1023;
    out[(size_t)b * 3072 + 1024 + e] = (s0 > 20.f) ? s0 : log1pf(__expf(s0));
    out[(size_t)b * 3072 + 2048 + e] = (s1 > 20.f) ? s1 : log1pf(__expf(s1));
}

// ---------------------------------------------------------------------------
// Host-side GEMM dispatch
// ---------------------------------------------------------------------------
static void gemm(hipStream_t s, const float* Aa, int lda, const float* Bw, int ldb,
                 const float* bias, float* Cc_, int ldc, int M, int N, int K,
                 bool accum, bool relu) {
    dim3 g((M + BM - 1) / BM, (N + BN - 1) / BN), b(256);
    bool full = (M % BM == 0) && (N % BN == 0) && (K % BK == 0);
    if (full) {
        if (accum)     gemm_wmma_bf16<true,  0, true><<<g, b, 0, s>>>(Aa, lda, Bw, ldb, bias, Cc_, ldc, M, N, K);
        else if (relu) gemm_wmma_bf16<false, 1, true><<<g, b, 0, s>>>(Aa, lda, Bw, ldb, bias, Cc_, ldc, M, N, K);
        else           gemm_wmma_bf16<false, 0, true><<<g, b, 0, s>>>(Aa, lda, Bw, ldb, bias, Cc_, ldc, M, N, K);
    } else {
        if (accum)     gemm_wmma_bf16<true,  0, false><<<g, b, 0, s>>>(Aa, lda, Bw, ldb, bias, Cc_, ldc, M, N, K);
        else if (relu) gemm_wmma_bf16<false, 1, false><<<g, b, 0, s>>>(Aa, lda, Bw, ldb, bias, Cc_, ldc, M, N, K);
        else           gemm_wmma_bf16<false, 0, false><<<g, b, 0, s>>>(Aa, lda, Bw, ldb, bias, Cc_, ldc, M, N, K);
    }
}

#define GRID1(n) dim3(((n) + 255) / 256), dim3(256)

extern "C" void kernel_launch(void* const* d_in, const int* in_sizes, int n_in,
                              void* d_out, int out_size, void* d_ws, size_t ws_size,
                              hipStream_t stream) {
    (void)in_sizes; (void)n_in; (void)out_size; (void)ws_size;
    const float* v_mag    = (const float*)d_in[0];
    const float* p_bus    = (const float*)d_in[1];
    const float* q_bus    = (const float*)d_in[2];
    const float* obs_mask = (const float*)d_in[3];
    const int*   ei       = (const int*)  d_in[4];
    const float* w_in     = (const float*)d_in[5];
    const float* b_in     = (const float*)d_in[6];
    const float* gat0_lin = (const float*)d_in[7];
    const float* gat0_as  = (const float*)d_in[8];
    const float* gat0_ad  = (const float*)d_in[9];
    const float* gat0_b   = (const float*)d_in[10];
    const float* gat1_lin = (const float*)d_in[11];
    const float* gat1_as  = (const float*)d_in[12];
    const float* gat1_ad  = (const float*)d_in[13];
    const float* gat1_b   = (const float*)d_in[14];
    const float* norm_g   = (const float*)d_in[15];
    const float* norm_b   = (const float*)d_in[16];
    const float* m_inproj = (const float*)d_in[17];
    const float* m_convw  = (const float*)d_in[18];
    const float* m_convb  = (const float*)d_in[19];
    const float* m_xproj  = (const float*)d_in[20];
    const float* m_dtw    = (const float*)d_in[21];
    const float* m_dtb    = (const float*)d_in[22];
    const float* m_Alog   = (const float*)d_in[23];
    const float* m_D      = (const float*)d_in[24];
    const float* m_outp   = (const float*)d_in[25];
    const float* sh_w1 = (const float*)d_in[26]; const float* sh_b1 = (const float*)d_in[27];
    const float* sh_g1 = (const float*)d_in[28]; const float* sh_bb1= (const float*)d_in[29];
    const float* sh_w2 = (const float*)d_in[30]; const float* sh_b2 = (const float*)d_in[31];
    const float* sh_g2 = (const float*)d_in[32]; const float* sh_bb2= (const float*)d_in[33];
    const float* sh_w3 = (const float*)d_in[34]; const float* sh_b3 = (const float*)d_in[35];
    const float* ph_ew = (const float*)d_in[36]; const float* ph_eb = (const float*)d_in[37];
    const float* ph_w1 = (const float*)d_in[38]; const float* ph_b1 = (const float*)d_in[39];
    const float* ph_g1 = (const float*)d_in[40]; const float* ph_bb1= (const float*)d_in[41];
    const float* ph_w2 = (const float*)d_in[42]; const float* ph_b2 = (const float*)d_in[43];
    const float* ph_g2 = (const float*)d_in[44]; const float* ph_bb2= (const float*)d_in[45];
    const float* ph_w3 = (const float*)d_in[46]; const float* ph_b3 = (const float*)d_in[47];
    float* out = (float*)d_out;

    // ---- workspace linear allocator (floats) ----
    float* ws = (float*)d_ws;
    size_t off = 0;
    auto WA = [&](size_t n) { float* p = ws + off; off += n; return p; };
    float* X    = WA((size_t)ROWS * Cc);
    float* XH   = WA((size_t)ROWS * DM);
    float* G0   = WA((size_t)ROWS * DM);
    float* G1   = WA((size_t)ROWS * DM);
    float* U    = WA((size_t)ROWS * DM);
    float* ES   = WA((size_t)ROWS * Hh);
    float* ED   = WA((size_t)ROWS * Hh);
    float* MM   = WA((size_t)ROWS * Hh);
    float* DEN  = WA((size_t)ROWS * Hh);
    float* EATT = WA((size_t)GG * ETOT * Hh);
    float* TA   = WA((size_t)ROWS * DI);
    float* TB   = WA((size_t)ROWS * DI);
    float* TC   = WA((size_t)ROWS * DI);
    float* TD   = WA((size_t)ROWS * 48);
    float* POOL = WA((size_t)2048 * 512);
    float* EFP  = WA((size_t)2048 * 256);
    float* MT1  = WA((size_t)2048 * 128);
    float* MT2  = WA((size_t)2048 * 64);

    // ---- input projection ----
    k_input_proj<<<GRID1(ROWS * Cc), 0, stream>>>(v_mag, p_bus, q_bus, obs_mask, w_in, b_in, X);

    // ---- GAT layers (batched over all 128 graphs, one big GEMM each) ----
    auto run_gat = [&](const float* hin, int Kin, const float* lin, const float* asrc,
                       const float* adst, const float* bias, float* outbuf) {
        gemm(stream, hin, Kin, lin, DM, nullptr, XH, DM, ROWS, DM, Kin, false, false);
        k_nodesum<<<GRID1(ROWS * Hh), 0, stream>>>(XH, asrc, adst, ES, ED);
        k_fill<<<GRID1(ROWS * Hh), 0, stream>>>(MM, -3.0e38f, ROWS * Hh);
        k_fill<<<GRID1(ROWS * Hh), 0, stream>>>(DEN, 0.f, ROWS * Hh);
        k_edge_max<<<GRID1(GG * ETOT * Hh), 0, stream>>>(ei, ES, ED, EATT, MM);
        k_edge_expsum<<<GRID1(GG * ETOT * Hh), 0, stream>>>(ei, EATT, MM, DEN);
        k_init_bias256<<<GRID1(ROWS * DM), 0, stream>>>(outbuf, bias, ROWS * DM);
        k_edge_agg<<<dim3(GG * ETOT), dim3(256), 0, stream>>>(ei, EATT, DEN, XH, outbuf);
        k_elu<<<GRID1(ROWS * DM), 0, stream>>>(outbuf, ROWS * DM);
    };
    run_gat(X,  Cc, gat0_lin, gat0_as, gat0_ad, gat0_b, G0);
    run_gat(G0, DM, gat1_lin, gat1_as, gat1_ad, gat1_b, G1);

    // ---- (B,T,N,DM) -> (B,N,T,DM) residual stream ----
    k_transpose<<<GRID1(ROWS * DM), 0, stream>>>(G1, U);

    // ---- Mamba layers ----
    for (int i = 0; i < 3; ++i) {
        k_layernorm<<<dim3(ROWS), dim3(DM), DM * 4, stream>>>(U, G0, norm_g, norm_b, DM);
        const float* Wip = m_inproj + (size_t)i * DM * (2 * DI);
        gemm(stream, G0, DM, Wip,        2 * DI, nullptr, TA, DI, ROWS, DI, DM, false, false); // x branch
        gemm(stream, G0, DM, Wip + DI,   2 * DI, nullptr, TB, DI, ROWS, DI, DM, false, false); // z branch
        k_conv_silu<<<GRID1(ROWS * DI), 0, stream>>>(TA, m_convw + (size_t)i * DI * KK,
                                                     m_convb + (size_t)i * DI, TC);
        gemm(stream, TC, DI, m_xproj + (size_t)i * DI * 48, 48, nullptr, TD, 48, ROWS, 48, DI, false, false);
        gemm(stream, TD, 48, m_dtw + (size_t)i * DR * DI, DI, nullptr, TA, DI, ROWS, DI, DR, false, false);
        k_softplus_bias<<<GRID1(ROWS * DI), 0, stream>>>(TA, m_dtb + (size_t)i * DI, ROWS * DI);
        k_scan<<<dim3(Bb * Nn * 2), dim3(256), 0, stream>>>(TA, TD, TC, TB,
                                                            m_Alog + (size_t)i * DI * DSs,
                                                            m_D + (size_t)i * DI);
        gemm(stream, TC, DI, m_outp + (size_t)i * DI * DM, DM, nullptr, U, DM, ROWS, DM, DI, true, false);
    }

    // ---- state head: last timestep, strided-lda GEMM (no gather) ----
    gemm(stream, U + (size_t)(Tt - 1) * DM, Tt * DM, sh_w1, 128, sh_b1, MT1, 128, Bb * Nn, 128, DM, false, true);
    k_layernorm<<<dim3(Bb * Nn), dim3(128), 128 * 4, stream>>>(MT1, MT1, sh_g1, sh_bb1, 128);
    gemm(stream, MT1, 128, sh_w2, 64, sh_b2, MT2, 64, Bb * Nn, 64, 128, false, true);
    k_layernorm<<<dim3(Bb * Nn), dim3(64), 64 * 4, stream>>>(MT2, MT2, sh_g2, sh_bb2, 64);
    k_head_sh<<<GRID1(Bb * Nn), 0, stream>>>(MT2, sh_w3, sh_b3, out);

    // ---- param head: pool over T first (linearity), then project ----
    k_pool<<<GRID1(Bb * Ee * 2 * DM), 0, stream>>>(ei, U, POOL);
    gemm(stream, POOL, 2 * DM, ph_ew, DM, ph_eb, EFP, DM, Bb * Ee, DM, 2 * DM, false, false);
    gemm(stream, EFP, DM, ph_w1, 128, ph_b1, MT1, 128, Bb * Ee, 128, DM, false, true);
    k_layernorm<<<dim3(Bb * Ee), dim3(128), 128 * 4, stream>>>(MT1, MT1, ph_g1, ph_bb1, 128);
    gemm(stream, MT1, 128, ph_w2, 64, ph_b2, MT2, 64, Bb * Ee, 64, 128, false, true);
    k_layernorm<<<dim3(Bb * Ee), dim3(64), 64 * 4, stream>>>(MT2, MT2, ph_g2, ph_bb2, 64);
    k_head_ph<<<GRID1(Bb * Ee), 0, stream>>>(MT2, ph_w3, ph_b3, out);
}
